// Doc_Self_Atten_68616397521526
// MI455X (gfx1250) — matrix operand order repaired and checked by reading, not compiled, not run on hardware
//
#include <hip/hip_runtime.h>

// ---------------------------------------------------------------------------
// Doc self-attention: Out = softmax((D @ W^T) @ D^T) @ D
// B=8, N=2048, D_IN=D_HID=768.  All matmuls via v_wmma_f32_16x16x32_f16.
// ---------------------------------------------------------------------------

typedef __attribute__((ext_vector_type(16))) _Float16 v16h;
typedef __attribute__((ext_vector_type(8)))  _Float16 v8h;
typedef __attribute__((ext_vector_type(4)))  _Float16 v4h;
typedef __attribute__((ext_vector_type(8)))  float    v8f;

#define BATCH 8
#define SEQ   2048
#define DH    768
#define MBLK  32           // query rows per workgroup
#define KBLK  32           // keys per inner block
#define QLD   776          // padded f16 row stride for Krm (768+8 -> bank spread)
#define KCLD  40           // padded f16 row stride for Kcm/Pp (32+8)

// ---- WMMA fragment loaders (generic ptr: LDS or global) -------------------
// A (16x32 f16, MxK): lane<16 -> row=lane, K in {kb..kb+7, kb+16..kb+23}, kb = (lane>=16)*8
__device__ __forceinline__ v16h load_frag_a(const _Float16* p0, int ld) {
  int lane = threadIdx.x & 31;
  int row  = lane & 15;
  int kb   = (lane >> 4) << 3;           // 0 or 8
  const _Float16* p = p0 + row * ld + kb;
  v8h lo = *(const v8h*)(p);             // K = kb .. kb+7
  v8h hi = *(const v8h*)(p + 16);        // K = kb+16 .. kb+23
  return __builtin_shufflevector(lo, hi, 0,1,2,3,4,5,6,7,8,9,10,11,12,13,14,15);
}
// B (32x16 f16, KxN): lane<16 -> col=lane, K in {kb..kb+7, kb+16..kb+23}, kb = (lane>=16)*8
// Source tile must be laid out [N][K] row-major with stride ld (so K contiguous).
__device__ __forceinline__ v16h load_frag_b(const _Float16* p0, int ld) {
  int lane = threadIdx.x & 31;
  int col  = lane & 15;
  int kb   = (lane >> 4) << 3;           // 0 or 8
  const _Float16* p = p0 + col * ld + kb;
  v8h lo = *(const v8h*)(p);
  v8h hi = *(const v8h*)(p + 16);
  return __builtin_shufflevector(lo, hi, 0,1,2,3,4,5,6,7,8,9,10,11,12,13,14,15);
}
__device__ __forceinline__ v8f wmma_f16(v16h a, v16h b, v8f c) {
  return __builtin_amdgcn_wmma_f32_16x16x32_f16(false, a, false, b, (short)0, c, false, false);
}

// ---- f32 -> f16 conversion (vectorized) -----------------------------------
__global__ __launch_bounds__(256) void cvt_f32_f16(const float* __restrict__ src,
                                                   _Float16* __restrict__ dst, int n4) {
  int i = blockIdx.x * blockDim.x + threadIdx.x;
  if (i < n4) {
    float4 v = ((const float4*)src)[i];
    v4h h = { (_Float16)v.x, (_Float16)v.y, (_Float16)v.z, (_Float16)v.w };
    ((v4h*)dst)[i] = h;
  }
}

// ---- Projection: P[b,n,h] = sum_d D[b,n,d] * W[h,d]  (one 16x16 tile/wave) --
__global__ __launch_bounds__(128) void proj_kernel(const _Float16* __restrict__ D16,
                                                   const _Float16* __restrict__ W16,
                                                   _Float16* __restrict__ P16) {
  int tile = blockIdx.x * 4 + (threadIdx.x >> 5);   // tiles = B * (SEQ/16) * (DH/16)
  int ht   = tile % (DH / 16);
  int rest = tile / (DH / 16);
  int nt   = rest % (SEQ / 16);
  int b    = rest / (SEQ / 16);

  const _Float16* A  = D16 + ((size_t)b * SEQ + (size_t)nt * 16) * DH;
  const _Float16* Bm = W16 + (size_t)ht * 16 * DH;   // [h][d] rows: K (=d) contiguous
  v8f c = {};
#pragma unroll
  for (int ks = 0; ks < DH / 32; ++ks) {
    v16h a  = load_frag_a(A  + ks * 32, DH);
    v16h bm = load_frag_b(Bm + ks * 32, DH);
    c = wmma_f16(a, bm, c);
  }
  int lane = threadIdx.x & 31;
  int cn   = lane & 15;
  int cm0  = (lane >> 4) * 8;
  _Float16* Po = P16 + ((size_t)b * SEQ + (size_t)nt * 16) * DH + ht * 16;
#pragma unroll
  for (int g = 0; g < 8; ++g)
    Po[(size_t)(cm0 + g) * DH + cn] = (_Float16)c[g];
}

// ---- Flash attention: per workgroup, 32 query rows, online softmax ---------
__global__ __launch_bounds__(512, 1) void flash_kernel(const _Float16* __restrict__ Q16,
                                                       const _Float16* __restrict__ D16,
                                                       float* __restrict__ out) {
  extern __shared__ char smem_raw[];
  float*    O    = (float*)smem_raw;             // [MBLK][DH]
  float*    S    = O    + MBLK * DH;             // [MBLK][KBLK]
  float*    rowM = S    + MBLK * KBLK;           // [MBLK]
  float*    rowL = rowM + MBLK;                  // [MBLK]
  float*    rowA = rowL + MBLK;                  // [MBLK]
  _Float16* Krm  = (_Float16*)(rowA + MBLK);     // [KBLK][QLD]  keys x d
  _Float16* Kcm  = Krm + KBLK * QLD;             // [DH][KCLD]   d x keys
  _Float16* Pp   = Kcm + DH * KCLD;              // [MBLK][KCLD] probs (f16)

  int b   = blockIdx.y;
  int q0  = blockIdx.x * MBLK;
  int tid = threadIdx.x;
  int wave = tid >> 5;
  int lane = tid & 31;
  int cn   = lane & 15;
  int cm0  = (lane >> 4) * 8;

  const _Float16* Qg = Q16 + ((size_t)b * SEQ + q0) * DH;
  const _Float16* Dg = D16 + (size_t)b * SEQ * DH;

  // S-phase role of this wave is fixed for the whole kernel:
  int t  = wave & 3, kg = wave >> 2;   // tile = (row-tile rt, key-col-tile kt), k-slice kg
  int rt = t >> 1, kt = t & 1;

  // --- preload this wave's 6 loop-invariant Q A-fragments into VGPRs -------
  // (from global P16; each covers K = kg*192 + ks*32 .. +31 of rows rt*16..+15)
  v16h qf[6];
#pragma unroll
  for (int ks = 0; ks < 6; ++ks)
    qf[ks] = load_frag_a(Qg + (size_t)rt * 16 * DH + kg * 192 + ks * 32, DH);

  // --- init: zero O/S, init row stats ---
  for (int i = tid; i < MBLK * DH; i += 512) O[i] = 0.0f;
  for (int i = tid; i < MBLK * KBLK; i += 512) S[i] = 0.0f;
  if (tid < MBLK) { rowM[tid] = -1e30f; rowL[tid] = 0.0f; }
  __syncthreads();

  for (int kb = 0; kb < SEQ / KBLK; ++kb) {
    const _Float16* Kg = Dg + (size_t)kb * KBLK * DH;

    // --- stage key tile: row-major (keys x d) and d-major (d x keys) ---
    for (int c = tid; c < KBLK * (DH / 8); c += 512) {
      int key = c / (DH / 8), h0 = (c % (DH / 8)) * 8;
      v8h v = *(const v8h*)(Kg + key * DH + h0);
      *(v8h*)(Krm + key * QLD + h0) = v;
#pragma unroll
      for (int j = 0; j < 8; ++j) Kcm[(h0 + j) * KCLD + key] = v[j];
    }
    // --- prefetch next key tile into cache (global_prefetch_b8) ---
    if (kb + 1 < SEQ / KBLK) {
      const char* Kn = (const char*)(Kg + (size_t)KBLK * DH);
      for (int i = tid; i < (KBLK * DH * 2) / 128; i += 512)   // 384 cache lines
        __builtin_prefetch(Kn + i * 128, 0, 0);
    }
    __syncthreads();

    // --- S = Q K^T : 4 tiles (2 row x 2 col), 4 waves/tile x 6 k-steps ---
    {
      const _Float16* Bbase = Krm + kt * 16 * QLD + kg * 192;
      v8f c = {};
#pragma unroll
      for (int ks = 0; ks < 6; ++ks) {
        v16h bm = load_frag_b(Bbase + ks * 32, QLD);
        c = wmma_f16(qf[ks], bm, c);
      }
#pragma unroll
      for (int g = 0; g < 8; ++g)
        atomicAdd(&S[(rt * 16 + cm0 + g) * KBLK + kt * 16 + cn], c[g]);  // ds_add_f32
    }
    __syncthreads();

    // --- online softmax over this key block: wave owns rows 2w, 2w+1 ---
#pragma unroll
    for (int rr = 0; rr < 2; ++rr) {
      int r = wave * 2 + rr;
      float s = S[r * KBLK + lane];
      S[r * KBLK + lane] = 0.0f;                 // re-zero for next block
      float mx = s;
#pragma unroll
      for (int off = 16; off >= 1; off >>= 1) mx = fmaxf(mx, __shfl_xor(mx, off));
      float mOld = rowM[r];
      float mNew = fmaxf(mOld, mx);
      float p = __expf(s - mNew);
      float sum = p;
#pragma unroll
      for (int off = 16; off >= 1; off >>= 1) sum += __shfl_xor(sum, off);
      Pp[r * KCLD + lane] = (_Float16)p;
      if (lane == 0) {
        float alpha = __expf(mOld - mNew);
        rowA[r] = alpha;
        rowM[r] = mNew;
        rowL[r] = rowL[r] * alpha + sum;
      }
    }
    __syncthreads();

    // --- O = O*alpha + P @ V : 96 tiles (2 row x 48 col), 6 tiles/wave ---
#pragma unroll
    for (int tt = 0; tt < 6; ++tt) {
      int idx = wave * 6 + tt;
      int prt = idx / 48, ct = idx % 48;
      v16h a  = load_frag_a(Pp  + prt * 16 * KCLD, KCLD);
      v16h bm = load_frag_b(Kcm + ct * 16 * KCLD, KCLD);
      v8f c = {};
      c = wmma_f16(a, bm, c);
#pragma unroll
      for (int g = 0; g < 8; ++g) {
        int row = prt * 16 + cm0 + g;
        float* o = &O[row * DH + ct * 16 + cn];
        *o = *o * rowA[row] + c[g];
      }
    }
    __syncthreads();
  }

  // --- epilogue: normalize by row sum, write f32 ---
  float* Og = out + ((size_t)b * SEQ + q0) * DH;
  for (int i = tid; i < MBLK * DH; i += 512) {
    int r = i / DH;
    Og[i] = O[i] / rowL[r];
  }
}

// ---------------------------------------------------------------------------
extern "C" void kernel_launch(void* const* d_in, const int* in_sizes, int n_in,
                              void* d_out, int out_size, void* d_ws, size_t ws_size,
                              hipStream_t stream) {
  (void)in_sizes; (void)n_in; (void)out_size; (void)ws_size;
  const float* D = (const float*)d_in[0];   // [8,2048,768] f32
  const float* W = (const float*)d_in[1];   // [768,768]    f32
  float* out = (float*)d_out;               // [8,2048,768] f32

  char* ws = (char*)d_ws;
  constexpr size_t D_BYTES = (size_t)BATCH * SEQ * DH * 2;   // 25165824
  _Float16* D16 = (_Float16*)(ws);
  _Float16* P16 = (_Float16*)(ws + D_BYTES);
  _Float16* W16 = (_Float16*)(ws + 2 * D_BYTES);

  // f32 -> f16
  {
    int n4 = BATCH * SEQ * DH / 4;                  // 3145728
    cvt_f32_f16<<<(n4 + 255) / 256, 256, 0, stream>>>(D, D16, n4);
  }
  {
    int n4 = DH * DH / 4;                           // 147456
    cvt_f32_f16<<<(n4 + 255) / 256, 256, 0, stream>>>(W, W16, n4);
  }

  // P = D @ W^T  (f16 in, f32 acc, f16 out)
  {
    int tiles = BATCH * (SEQ / 16) * (DH / 16);     // 49152, 4 tiles/block
    proj_kernel<<<tiles / 4, 128, 0, stream>>>(D16, W16, P16);
  }

  // flash attention
  {
    constexpr int SMEM =
        (MBLK * DH + MBLK * KBLK + 3 * MBLK) * 4 +              // f32 region
        (KBLK * QLD + DH * KCLD + MBLK * KCLD) * 2;             // f16 region
    static_assert(SMEM <= 320 * 1024, "LDS budget");
    (void)hipFuncSetAttribute((const void*)flash_kernel,
                              hipFuncAttributeMaxDynamicSharedMemorySize, SMEM);
    flash_kernel<<<dim3(SEQ / MBLK, BATCH), 512, SMEM, stream>>>(P16, D16, out);
  }
}
